// SemanticLinkingHead_27960237097198
// MI455X (gfx1250) — matrix operand order repaired and checked by reading, not compiled, not run on hardware
//
#include <hip/hip_runtime.h>
#include <hip/hip_bf16.h>
#include <math.h>

// ---------------------------------------------------------------------------
// Problem constants (from reference)
// ---------------------------------------------------------------------------
#define N_CAND 32768
#define HIDDEN 1024
#define N_REL  10

// GEMM1 tiling
#define BM 128
#define BN 128
#define BK 32     // K-chunk per feature segment; 4 segments -> 128 effective K/iter
#define NKT (HIDDEN / BK)   // 32 k-iterations

typedef __attribute__((ext_vector_type(16))) __bf16 v16bf;
typedef __attribute__((ext_vector_type(8)))  __bf16 v8bf;
typedef __attribute__((ext_vector_type(8)))  float  v8f;

// Packed-W1 tile image: Bt[(nb*NKT + kt)*4 + seg][n(128)][k(32)] bf16, each
// (nb,kt) group = 32 KB contiguous == exact LDS image consumed per k-iter.
#define TILE_ELEMS (BN * BK)                    // 4096 bf16 per (seg) tile

// ---------------------------------------------------------------------------
// Stage 0: pack W1 (f32 [4096][1024]) -> bf16 pre-swizzled tiles (8 MiB).
// Runs once per launch; L2-resident afterwards. One block per (nb, kt, seg).
// ---------------------------------------------------------------------------
__global__ __launch_bounds__(256)
void w1_pack_kernel(const float* __restrict__ W1, __bf16* __restrict__ Bt)
{
    __shared__ __align__(16) __bf16 tile[BN][BK];   // 8 KB, [n][k]
    const int nb  = blockIdx.x;      // 0..7   n-block
    const int kt  = blockIdx.y;      // 0..31  k-chunk
    const int seg = blockIdx.z;      // 0..3   feature segment
    const int t   = threadIdx.x;

    const int kq = t >> 5;           // 0..7
    const int n4 = (t & 31) * 4;     // 0..124
#pragma unroll
    for (int p = 0; p < 4; ++p) {
        const int k = p * 8 + kq;
        const float4 w = *(const float4*)(W1 + (size_t)(seg * HIDDEN + kt * BK + k) * HIDDEN + nb * BN + n4);
        tile[n4 + 0][k] = (__bf16)w.x;
        tile[n4 + 1][k] = (__bf16)w.y;
        tile[n4 + 2][k] = (__bf16)w.z;
        tile[n4 + 3][k] = (__bf16)w.w;
    }
    __syncthreads();

    __bf16* dst = Bt + (size_t)((nb * NKT + kt) * 4 + seg) * TILE_ELEMS;
    const uint4* s4 = (const uint4*)&tile[0][0];
    uint4* d4 = (uint4*)dst;
#pragma unroll
    for (int i = 0; i < 2; ++i)      // 8 KB / 16 B / 256 thr = 2
        d4[t + i * 256] = s4[t + i * 256];
}

// ---------------------------------------------------------------------------
// Stage 1: fused gather + feature-build + (feats @ W1 + b1) + exact GELU.
// Register-double-buffered: next iter's global loads issue under the WMMAs.
// ---------------------------------------------------------------------------
__global__ __launch_bounds__(256)
void gemm1_gelu_kernel(const float* __restrict__ emb,
                       const int*   __restrict__ src_idx,
                       const int*   __restrict__ dst_idx,
                       const __bf16* __restrict__ Bt,
                       const float* __restrict__ b1,
                       float*       __restrict__ h_out)
{
    __shared__ __align__(64) __bf16 Alds[4][BM][BK];   // 32 KB, [seg][m][k]
    __shared__ __align__(64) __bf16 Blds[4][BN][BK];   // 32 KB, [seg][n][k]
    __shared__ int sIdx[BM];
    __shared__ int dIdx[BM];

    const int t  = threadIdx.x;
    const int m0 = blockIdx.x * BM;
    const int nb = blockIdx.y;
    const int n0 = nb * BN;

    if (t < BM)      sIdx[t]      = src_idx[m0 + t];
    else             dIdx[t - BM] = dst_idx[m0 + (t - BM)];
    __syncthreads();

    // wave mapping: 8 waves cover 128x128; each wave owns 32(m) x 64(n)
    const int lane = t & 31;
    const int wave = t >> 5;
    const int wm   = (wave & 3) * 32;
    const int wn   = (wave >> 2) * 64;
    const int lr   = lane & 15;
    const int hi   = lane >> 4;

    v8f acc[2][4];
#pragma unroll
    for (int mi = 0; mi < 2; ++mi)
#pragma unroll
        for (int ni = 0; ni < 4; ++ni) acc[mi][ni] = (v8f)0.0f;

    // A global-load mapping: 4 passes of (32 m-rows x 32 k) float4 per thread
    const int am = t >> 3;
    const int ak = (t & 7) * 4;
    const float* srow[4];
    const float* drow[4];
#pragma unroll
    for (int p = 0; p < 4; ++p) {
        srow[p] = emb + (size_t)sIdx[p * 32 + am] * HIDDEN + ak;
        drow[p] = emb + (size_t)dIdx[p * 32 + am] * HIDDEN + ak;
    }

    // B copy source for this n-block (32 KB contiguous per k-iter)
    const uint4* bsrc0 = (const uint4*)(Bt + (size_t)(nb * NKT) * 4 * TILE_ELEMS);
    uint4* bdst = (uint4*)&Blds[0][0][0];

    // ---- prefetch registers ----
    float4 sreg[4], dreg[4];
    uint4  breg[8];

    // preload iteration 0
#pragma unroll
    for (int p = 0; p < 4; ++p) { sreg[p] = *(const float4*)(srow[p]); dreg[p] = *(const float4*)(drow[p]); }
#pragma unroll
    for (int i = 0; i < 8; ++i) breg[i] = bsrc0[t + i * 256];

    for (int kt = 0; kt < NKT; ++kt) {
        // ---- commit prefetched data to LDS ----
#pragma unroll
        for (int p = 0; p < 4; ++p) {
            const int m = p * 32 + am;
            const float4 s = sreg[p];
            const float4 d = dreg[p];
            __bf16* a0 = &Alds[0][m][ak];
            __bf16* a1 = &Alds[1][m][ak];
            __bf16* a2 = &Alds[2][m][ak];
            __bf16* a3 = &Alds[3][m][ak];
            a0[0]=(__bf16)s.x; a0[1]=(__bf16)s.y; a0[2]=(__bf16)s.z; a0[3]=(__bf16)s.w;
            a1[0]=(__bf16)d.x; a1[1]=(__bf16)d.y; a1[2]=(__bf16)d.z; a1[3]=(__bf16)d.w;
            a2[0]=(__bf16)fabsf(s.x-d.x); a2[1]=(__bf16)fabsf(s.y-d.y);
            a2[2]=(__bf16)fabsf(s.z-d.z); a2[3]=(__bf16)fabsf(s.w-d.w);
            a3[0]=(__bf16)(s.x*d.x); a3[1]=(__bf16)(s.y*d.y);
            a3[2]=(__bf16)(s.z*d.z); a3[3]=(__bf16)(s.w*d.w);
        }
#pragma unroll
        for (int i = 0; i < 8; ++i) bdst[t + i * 256] = breg[i];
        __syncthreads();

        // ---- issue next iteration's global loads (latency hides under WMMA)
        if (kt + 1 < NKT) {
            const int k0n = (kt + 1) * BK;
            const uint4* bsrc = bsrc0 + (size_t)(kt + 1) * (4 * TILE_ELEMS / 8);
#pragma unroll
            for (int p = 0; p < 4; ++p) {
                sreg[p] = *(const float4*)(srow[p] + k0n);
                dreg[p] = *(const float4*)(drow[p] + k0n);
            }
#pragma unroll
            for (int i = 0; i < 8; ++i) breg[i] = bsrc[t + i * 256];
        }

        // ---- WMMA: 4 segments x 2 M-frags x 4 N-frags ----
#pragma unroll
        for (int s = 0; s < 4; ++s) {
            v16bf afrag[2];
#pragma unroll
            for (int mi = 0; mi < 2; ++mi) {
                // A 16x32 layout: lane<16 holds K 0-15, lane>=16 holds K 16-31 of row lr
                const __bf16* arow = &Alds[s][wm + mi * 16 + lr][0];
                v8bf lo = *(const v8bf*)(arow + hi * 16);
                v8bf up = *(const v8bf*)(arow + hi * 16 + 8);
                v16bf a;
#pragma unroll
                for (int e = 0; e < 8; ++e) { a[e] = lo[e]; a[e + 8] = up[e]; }
                afrag[mi] = a;
            }
#pragma unroll
            for (int ni = 0; ni < 4; ++ni) {
                // B 32x16: lane<16 -> col lr K 0-15; lane>=16 -> col lr K 16-31
                const __bf16* bcol = &Blds[s][wn + ni * 16 + lr][0];
                v16bf b = *(const v16bf*)(bcol + hi * 16);
#pragma unroll
                for (int mi = 0; mi < 2; ++mi) {
                    acc[mi][ni] = __builtin_amdgcn_wmma_f32_16x16x32_bf16(
                        false, afrag[mi], false, b, (short)0, acc[mi][ni], false, false);
                }
            }
        }
        __syncthreads();
    }

    // ---- epilogue: + b1, exact GELU, store h (f32) ----
#pragma unroll
    for (int ni = 0; ni < 4; ++ni) {
        const int n = n0 + wn + ni * 16 + lr;
        const float bias = b1[n];
#pragma unroll
        for (int mi = 0; mi < 2; ++mi) {
#pragma unroll
            for (int r = 0; r < 8; ++r) {
                const int m = m0 + wm + mi * 16 + r + hi * 8;  // C/D: VGPR r -> M=r / 8+r
                const float x = acc[mi][ni][r] + bias;
                const float g = 0.5f * x * (1.0f + erff(x * 0.70710678118654752f));
                h_out[(size_t)m * HIDDEN + n] = g;
            }
        }
    }
}

// ---------------------------------------------------------------------------
// Stage 2: logits = h @ W2 + b2 ; log-softmax ; per-candidate NLL term.
// ---------------------------------------------------------------------------
__global__ __launch_bounds__(256)
void head_kernel(const float* __restrict__ h,
                 const float* __restrict__ W2,
                 const float* __restrict__ b2,
                 const int*   __restrict__ labels,
                 float*       __restrict__ logits_out,
                 float*       __restrict__ loss_terms)
{
    const int lane = threadIdx.x & 31;
    const int wave = threadIdx.x >> 5;
    const int c    = blockIdx.x * 8 + wave;

    const float* hc = h + (size_t)c * HIDDEN;
    float acc[N_REL];
#pragma unroll
    for (int j = 0; j < N_REL; ++j) acc[j] = 0.0f;

    for (int k = lane; k < HIDDEN; k += 32) {
        const float hv = hc[k];
        const float* w = W2 + (size_t)k * N_REL;
#pragma unroll
        for (int j = 0; j < N_REL; ++j) acc[j] = fmaf(hv, w[j], acc[j]);
    }
#pragma unroll
    for (int j = 0; j < N_REL; ++j)
        for (int off = 16; off >= 1; off >>= 1)
            acc[j] += __shfl_xor(acc[j], off, 32);

    if (lane == 0) {
        float lg[N_REL], mx = -INFINITY;
#pragma unroll
        for (int j = 0; j < N_REL; ++j) { lg[j] = acc[j] + b2[j]; mx = fmaxf(mx, lg[j]); }
        float se = 0.0f;
#pragma unroll
        for (int j = 0; j < N_REL; ++j) se += expf(lg[j] - mx);
        const float lse = mx + logf(se);
#pragma unroll
        for (int j = 0; j < N_REL; ++j) logits_out[(size_t)c * N_REL + j] = lg[j];
        loss_terms[c] = lse - lg[labels[c]];   // -log p[label]
    }
}

// ---------------------------------------------------------------------------
// Stage 3: deterministic mean reduction of loss terms -> d_out[0]
// ---------------------------------------------------------------------------
__global__ __launch_bounds__(256)
void loss_reduce_kernel(const float* __restrict__ loss_terms, float* __restrict__ out)
{
    __shared__ float red[256];
    float s = 0.0f;
    for (int i = threadIdx.x; i < N_CAND; i += 256) s += loss_terms[i];
    red[threadIdx.x] = s;
    __syncthreads();
    for (int st = 128; st > 0; st >>= 1) {
        if (threadIdx.x < st) red[threadIdx.x] += red[threadIdx.x + st];
        __syncthreads();
    }
    if (threadIdx.x == 0) out[0] = red[0] * (1.0f / (float)N_CAND);
}

// ---------------------------------------------------------------------------
// Launch. Inputs: node_embeddings, src_idx, dst_idx, labels, W1, b1, W2, b2.
// Output: [loss, logits(32768x10)] flat.
// Workspace: [0, 8MiB) packed W1 bf16 ; [8MiB, 136MiB) h f32 ; then loss terms.
// ---------------------------------------------------------------------------
extern "C" void kernel_launch(void* const* d_in, const int* in_sizes, int n_in,
                              void* d_out, int out_size, void* d_ws, size_t ws_size,
                              hipStream_t stream)
{
    const float* emb    = (const float*)d_in[0];
    const int*   src    = (const int*)  d_in[1];
    const int*   dst    = (const int*)  d_in[2];
    const int*   labels = (const int*)  d_in[3];
    const float* W1     = (const float*)d_in[4];
    const float* b1     = (const float*)d_in[5];
    const float* W2     = (const float*)d_in[6];
    const float* b2     = (const float*)d_in[7];

    float* out = (float*)d_out;                 // out[0]=loss, out[1..]=logits

    const size_t btBytes = (size_t)4 * HIDDEN * HIDDEN * sizeof(__bf16);   // 8 MiB
    __bf16* Bt = (__bf16*)d_ws;
    float*  h  = (float*)((char*)d_ws + btBytes);
    float*  lt = (float*)((char*)d_ws + btBytes + (size_t)N_CAND * HIDDEN * sizeof(float));

    w1_pack_kernel<<<dim3(HIDDEN / BN, NKT, 4), dim3(256), 0, stream>>>(W1, Bt);

    dim3 g1(N_CAND / BM, HIDDEN / BN);          // 256 x 8 blocks
    gemm1_gelu_kernel<<<g1, dim3(256), 0, stream>>>(emb, src, dst, Bt, b1, h);
    head_kernel<<<N_CAND / 8, 256, 0, stream>>>(h, W2, b2, labels, out + 1, lt);
    loss_reduce_kernel<<<1, 256, 0, stream>>>(lt, out);
}
